// RMatrixNet_56272661512383
// MI455X (gfx1250) — matrix operand order repaired and checked by reading, not compiled, not run on hardware
//
#include <hip/hip_runtime.h>

typedef __attribute__((ext_vector_type(16))) _Float16 v16h;
typedef __attribute__((ext_vector_type(8)))  float    v8f;

#define NMLP   256     // MLPs per branch (N*N)
#define WID    50      // hidden width
#define WPAD   64      // padded width (4 x 16)
#define BATCH  4096

__device__ __forceinline__ float swishf(float x) {
    // swish(x) = x*sigmoid(x) = 0.5x*tanh(x/2) + 0.5x  (3 ops, 1 TRANS on gfx1250)
    // swish(0) == 0 so 50->64 zero padding stays exact.
#if __has_builtin(__builtin_amdgcn_tanhf)
    const float hx = 0.5f * x;
    return fmaf(hx, __builtin_amdgcn_tanhf(hx), hx);
#else
    return x * __builtin_amdgcn_rcpf(1.0f + __expf(-x));
#endif
}

// xor-butterfly step within each 16-lane row: pure VALU v_permlane16_b32
// (no LDS round trip / s_wait_dscnt, unlike ds_bpermute-backed __shfl_xor).
__device__ __forceinline__ float row16_xor(float x, unsigned sel_lo, unsigned sel_hi, int xmask) {
#if __has_builtin(__builtin_amdgcn_permlane16)
    (void)xmask;
    union { float f; int i; } a, r;
    a.f = x;
    r.i = __builtin_amdgcn_permlane16(a.i, a.i, (int)sel_lo, (int)sel_hi, false, false);
    return r.f;
#else
    (void)sel_lo; (void)sel_hi;
    return __shfl_xor(x, xmask, 32);
#endif
}

__global__ __launch_bounds__(256, 1) void rmatrixnet_wmma(
    const float* __restrict__ u,
    const float* __restrict__ W1a, const float* __restrict__ b1a,
    const float* __restrict__ W2a, const float* __restrict__ b2a,
    const float* __restrict__ W3a, const float* __restrict__ b3a,
    const float* __restrict__ W1b, const float* __restrict__ b1b,
    const float* __restrict__ W2b, const float* __restrict__ b2b,
    const float* __restrict__ W3b, const float* __restrict__ b3b,
    float* __restrict__ out)
{
    // LDS: W2 as padded f16 (8KB) + full u vector (16KB) + small padded vectors (1KB)
    __shared__ __align__(32) _Float16 sW2[WPAD][WPAD];
    __shared__ float sU[BATCH];
    __shared__ float sW1[WPAD], sB1[WPAD], sB2[WPAD], sW3[WPAD];

    const int tid    = threadIdx.x;
    const int branch = blockIdx.x & 1;   // 0 = real, 1 = imag
    const int m      = blockIdx.x >> 1;  // which scalar MLP

    const float* W1 = branch ? W1b : W1a;
    const float* B1 = branch ? b1b : b1a;
    const float* W2 = branch ? W2b : W2a;
    const float* B2 = branch ? b2b : b2a;
    const float* W3 = branch ? W3b : W3a;
    const float  b3v = (branch ? b3b : b3a)[m];

    // ---- stage weights/inputs into LDS (zero-padded 50 -> 64) ----
    for (int i = tid; i < BATCH; i += 256) sU[i] = u[i];
    if (tid < WPAD) {
        const bool ok = tid < WID;
        sW1[tid] = ok ? W1[m * WID + tid] : 0.0f;
        sB1[tid] = ok ? B1[m * WID + tid] : 0.0f;
        sB2[tid] = ok ? B2[m * WID + tid] : 0.0f;
        sW3[tid] = ok ? W3[m * WID + tid] : 0.0f;
    }
    for (int i = tid; i < WPAD * WPAD; i += 256) {
        const int v = i >> 6, w = i & 63;
        const float x = (v < WID && w < WID) ? W2[(m * WID + v) * WID + w] : 0.0f;
        sW2[v][w] = (_Float16)x;
    }
    __syncthreads();

    const int lane = tid & 31;
    const int wv   = tid >> 5;       // wave id 0..7
    const int hsel = lane >> 4;      // half-wave select (K-split per ISA layout)
    const int lcol = lane & 15;      // N / M index within 16

    // ---- batch-invariant B fragments: h2[b,v] = sum_w h1[b,w] * W2[v,w]
    // B-matrix element (K=w, N=v) = W2[v][w]; lane holds fixed N=lcol,
    // K = kchunk*32 + hsel*16 + 0..15 -> 32 contiguous bytes in sW2 row.
    v16h bf[4][2];
#pragma unroll
    for (int n = 0; n < 4; ++n)
#pragma unroll
        for (int k = 0; k < 2; ++k)
            bf[n][k] = *(const v16h*)&sW2[n * 16 + lcol][k * 32 + hsel * 16];

    // ---- main loop: each wave owns batch tiles wv, wv+8, ... (32 tiles each) ----
    for (int t = wv; t < BATCH / 16; t += 8) {
        const float uv = sU[t * 16 + lcol];

        // Layer 1 computed straight into WMMA A-fragment layout:
        // af[k][h]: K = k*32 + hsel*16 + h
        v16h af[2];
#pragma unroll
        for (int k = 0; k < 2; ++k) {
#pragma unroll
            for (int h = 0; h < 16; ++h) {
                const int col = k * 32 + hsel * 16 + h;
                const float x = fmaf(uv, sW1[col], sB1[col]);
                af[k][h] = (_Float16)swishf(x);
            }
        }

        // 4 N-tiles x (K=64 as two K32 steps) = 8 WMMAs, fp32 accumulate
        v8f acc[4];
#pragma unroll
        for (int n = 0; n < 4; ++n) {
            v8f z = {};
            acc[n] = z;
            acc[n] = __builtin_amdgcn_wmma_f32_16x16x32_f16(
                false, af[0], false, bf[n][0], (short)0, acc[n], false, false);
            acc[n] = __builtin_amdgcn_wmma_f32_16x16x32_f16(
                false, af[1], false, bf[n][1], (short)0, acc[n], false, false);
        }

        // Epilogue: h2 = swish(acc + b2); out = <h2, W3> + b3.
        // C layout: acc[n][j] @ lane L -> row = t*16 + j + (L>=16 ? 8 : 0), v = n*16 + L%16
        float partial[8];
#pragma unroll
        for (int j = 0; j < 8; ++j) {
            float p = 0.0f;
#pragma unroll
            for (int n = 0; n < 4; ++n) {
                const int v = n * 16 + lcol;
                const float h2 = swishf(acc[n][j] + sB2[v]);
                p = fmaf(h2, sW3[v], p);
            }
            partial[j] = p;
        }

        // Butterfly over the 16 N-lanes of each half-row: level-outer so all
        // 8 independent trees pipeline together (VALU permlane16, no DS waits).
#pragma unroll
        for (int j = 0; j < 8; ++j) partial[j] += row16_xor(partial[j], 0x67452301u, 0xEFCDAB89u, 1);
#pragma unroll
        for (int j = 0; j < 8; ++j) partial[j] += row16_xor(partial[j], 0x54761032u, 0xDCFE98BAu, 2);
#pragma unroll
        for (int j = 0; j < 8; ++j) partial[j] += row16_xor(partial[j], 0x32107654u, 0xBA98FEDCu, 4);
#pragma unroll
        for (int j = 0; j < 8; ++j) partial[j] += row16_xor(partial[j], 0xFEDCBA98u, 0x76543210u, 8);

        if (lcol == 0) {
#pragma unroll
            for (int j = 0; j < 8; ++j) {
                const int row = t * 16 + j + hsel * 8;
                // complex64 layout: interleaved (re, im) floats
                out[((size_t)row * NMLP + m) * 2 + branch] = partial[j] + b3v;
            }
        }
    }
}

extern "C" void kernel_launch(void* const* d_in, const int* in_sizes, int n_in,
                              void* d_out, int out_size, void* d_ws, size_t ws_size,
                              hipStream_t stream) {
    (void)in_sizes; (void)n_in; (void)out_size; (void)d_ws; (void)ws_size;
    const float* u   = (const float*)d_in[0];
    const float* W1r = (const float*)d_in[1];
    const float* b1r = (const float*)d_in[2];
    const float* W2r = (const float*)d_in[3];
    const float* b2r = (const float*)d_in[4];
    const float* W3r = (const float*)d_in[5];
    const float* b3r = (const float*)d_in[6];
    const float* W1i = (const float*)d_in[7];
    const float* b1i = (const float*)d_in[8];
    const float* W2i = (const float*)d_in[9];
    const float* b2i = (const float*)d_in[10];
    const float* W3i = (const float*)d_in[11];
    const float* b3i = (const float*)d_in[12];
    float* out = (float*)d_out;

    // 256 MLPs x 2 branches = 512 blocks, 8 waves each
    rmatrixnet_wmma<<<dim3(NMLP * 2), dim3(256), 0, stream>>>(
        u, W1r, b1r, W2r, b2r, W3r, b3r,
        W1i, b1i, W2i, b2i, W3i, b3i, out);
}
